// MultiHeadedAttention_30159260352759
// MI455X (gfx1250) — matrix operand order repaired and checked by reading, not compiled, not run on hardware
//
#include <hip/hip_runtime.h>
#include <stdint.h>

typedef __attribute__((ext_vector_type(16))) __bf16 v16bf;
typedef __attribute__((ext_vector_type(8)))  float  v8f;

#define WMMA_BF16(a, b, c) \
  __builtin_amdgcn_wmma_f32_16x16x32_bf16(false, (a), false, (b), (short)0, (c), false, false)

constexpr int Bb = 2, Ss = 2048, Dd = 1024, Hh = 16, DK = 64;
constexpr int Mrows = Bb * Ss;   // 4096
constexpr int MW    = Ss / 32;   // mask words per row = 64

union Frag16 { v16bf v; uint32_t u[8]; __bf16 h[16]; };
union Pack2  { __bf16 h[2]; uint32_t u; };

__device__ __forceinline__ __bf16 f2bf(float f) { return (__bf16)f; }

// ---------------------------------------------------------------- converts
// f32 -> bf16, 4 elements per thread (n divisible by 4)
__global__ void mha_cvt_bf16(const float* __restrict__ x, __bf16* __restrict__ y, int n4) {
  int i = blockIdx.x * blockDim.x + threadIdx.x;
  if (i < n4) {
    float4 f = ((const float4*)x)[i];
    __bf16* o = y + (size_t)i * 4;
    o[0] = (__bf16)f.x; o[1] = (__bf16)f.y; o[2] = (__bf16)f.z; o[3] = (__bf16)f.w;
  }
}

// WT[n*D + k] = W[k*D + n] (bf16), LDS-tiled 32x32 so both sides are coalesced
__global__ void mha_wtrans_bf16(const float* __restrict__ w, __bf16* __restrict__ wt) {
  __shared__ float tile[32][33];
  const int bx = blockIdx.x * 32;  // n range (read), n (row) offset on write side
  const int by = blockIdx.y * 32;  // k range (read)
  const int tx = threadIdx.x, ty = threadIdx.y;  // block (32,8)
#pragma unroll
  for (int j = 0; j < 32; j += 8)
    tile[ty + j][tx] = w[(size_t)(by + ty + j) * Dd + bx + tx];
  __syncthreads();
#pragma unroll
  for (int j = 0; j < 32; j += 8)
    wt[(size_t)(bx + ty + j) * Dd + by + tx] = (__bf16)tile[tx][ty + j];
}

// pack mask [B,S,S] i32 -> bit-per-key words [B,S,S/32]
__global__ void mha_maskpack(const int* __restrict__ mask, uint32_t* __restrict__ mp, int nwords) {
  int i = blockIdx.x * blockDim.x + threadIdx.x;
  if (i < nwords) {
    const int* src = mask + (size_t)i * 32;
    uint32_t m = 0;
#pragma unroll
    for (int j = 0; j < 32; j++) m |= (src[j] != 0 ? 1u : 0u) << j;
    mp[i] = m;
  }
}

// ---------------------------------------------------------------- GEMM
// C[M=4096,N=1024] = A[M,K] bf16 x BT[N,K] bf16 + bias, software-pipelined K loop.
// mode 0: dstB per-head [B,H,S,DK] ; mode 1: dstB per-head [B,H,DK,S] ; mode 2: dstF f32 [M,N]
struct StepFrags { Frag16 a, b0, b1, b2, b3; };

__global__ void mha_gemm_bf16(const __bf16* __restrict__ A, const __bf16* __restrict__ BT,
                              const float* __restrict__ bias, __bf16* __restrict__ dstB,
                              float* __restrict__ dstF, int mode) {
  const int lane  = threadIdx.x & 31;
  const int wave  = threadIdx.x >> 5;
  const int mbase = blockIdx.x * 64 + wave * 16;
  const int nbase = blockIdx.y * 64;
  const int half  = (lane >> 4) & 1;
  const int nl    = lane & 15;

  const __bf16* Arow = A + (size_t)(mbase + nl) * Dd;
  const __bf16* Bp0  = BT + (size_t)(nbase + 0  + nl) * Dd + half * 16;
  const __bf16* Bp1  = BT + (size_t)(nbase + 16 + nl) * Dd + half * 16;
  const __bf16* Bp2  = BT + (size_t)(nbase + 32 + nl) * Dd + half * 16;
  const __bf16* Bp3  = BT + (size_t)(nbase + 48 + nl) * Dd + half * 16;

  auto load_step = [&](int k, StepFrags& F) {
#pragma unroll
    for (int i = 0; i < 8; i++) {
      int koff = k + half * 16 + 2 * i;
      F.a.u[i] = *(const uint32_t*)(Arow + koff);
    }
#pragma unroll
    for (int j = 0; j < 8; j++) {
      F.b0.u[j] = *(const uint32_t*)(Bp0 + k + 2 * j);
      F.b1.u[j] = *(const uint32_t*)(Bp1 + k + 2 * j);
      F.b2.u[j] = *(const uint32_t*)(Bp2 + k + 2 * j);
      F.b3.u[j] = *(const uint32_t*)(Bp3 + k + 2 * j);
    }
  };

  v8f acc0 = {}, acc1 = {}, acc2 = {}, acc3 = {};
  StepFrags F[2];
  load_step(0, F[0]);
#pragma unroll 4
  for (int it = 0; it < 31; ++it) {
    load_step((it + 1) * 32, F[(it + 1) & 1]);   // prefetch next step into other buffer
    StepFrags& C = F[it & 1];
    acc0 = WMMA_BF16(C.a.v, C.b0.v, acc0);
    acc1 = WMMA_BF16(C.a.v, C.b1.v, acc1);
    acc2 = WMMA_BF16(C.a.v, C.b2.v, acc2);
    acc3 = WMMA_BF16(C.a.v, C.b3.v, acc3);
  }
  {
    StepFrags& C = F[31 & 1];
    acc0 = WMMA_BF16(C.a.v, C.b0.v, acc0);
    acc1 = WMMA_BF16(C.a.v, C.b1.v, acc1);
    acc2 = WMMA_BF16(C.a.v, C.b2.v, acc2);
    acc3 = WMMA_BF16(C.a.v, C.b3.v, acc3);
  }

  v8f accs[4] = {acc0, acc1, acc2, acc3};
#pragma unroll
  for (int j = 0; j < 4; j++) {
#pragma unroll
    for (int r = 0; r < 8; r++) {
      int m = mbase + half * 8 + r;
      int n = nbase + j * 16 + nl;
      float val = accs[j][r] + bias[n];
      if (mode == 2) {
        dstF[(size_t)m * Dd + n] = val;
      } else {
        int bi = m / Ss, s = m % Ss;
        int h = n >> 6, dk = n & 63;
        if (mode == 0)
          dstB[(((size_t)bi * Hh + h) * Ss + s) * DK + dk] = f2bf(val);
        else
          dstB[(((size_t)bi * Hh + h) * DK + dk) * Ss + s] = f2bf(val);
      }
    }
  }
}

// ---------------------------------------------------------------- fused attention
// One wave per 16-row q tile; flash-style streaming, 32 keys/iter.
// Q,K: [B,H,S,DK] bf16 ; Vt: [B,H,DK,S] bf16 ; mp: packed mask bits ; out: [B,S,D] bf16
__global__ void mha_attn(const __bf16* __restrict__ Q, const __bf16* __restrict__ K,
                         const __bf16* __restrict__ Vt, const uint32_t* __restrict__ mp,
                         __bf16* __restrict__ attnOut) {
  __shared__ __bf16 Pbuf[4][32 * 16];  // per-wave, column-major [kcol][m]

  const int lane = threadIdx.x & 31;
  const int wave = threadIdx.x >> 5;
  const int tile = blockIdx.x * 4 + wave;
  const int qt   = tile & 127;          // S/16 = 128
  const int h    = (tile >> 7) & 15;
  const int bi   = tile >> 11;
  const int qbase = qt * 16;
  const int half  = (lane >> 4) & 1;
  const int nl    = lane & 15;

  const __bf16* Qh = Q  + ((size_t)bi * Hh + h) * Ss * DK;
  const __bf16* Kh = K  + ((size_t)bi * Hh + h) * Ss * DK;
  const __bf16* Vh = Vt + ((size_t)bi * Hh + h) * DK * Ss;

  // Q A-fragments (16x64 = two 16x32 fragments), constant over the k-loop
  Frag16 qa0, qa1;
  {
    const __bf16* Qrow = Qh + (size_t)(qbase + nl) * DK;
#pragma unroll
    for (int i = 0; i < 8; i++) {
      int koff = half * 16 + 2 * i;
      qa0.u[i] = *(const uint32_t*)(Qrow + koff);
      qa1.u[i] = *(const uint32_t*)(Qrow + 32 + koff);
    }
  }

  // per-row packed-mask word pointers (row = qbase + half*8 + r)
  const uint32_t* mrow[8];
#pragma unroll
  for (int r = 0; r < 8; r++)
    mrow[r] = mp + (size_t)(bi * Ss + qbase + half * 8 + r) * MW;

  float mrun[8], lrun[8];
  v8f o0 = {}, o1 = {}, o2 = {}, o3 = {};
#pragma unroll
  for (int r = 0; r < 8; r++) { mrun[r] = -1e30f; lrun[r] = 0.f; }

  __bf16* Pw = &Pbuf[wave][0];
  const __bf16* V0 = Vh + (size_t)(0  + nl) * Ss + half * 16;
  const __bf16* V1 = Vh + (size_t)(16 + nl) * Ss + half * 16;
  const __bf16* V2 = Vh + (size_t)(32 + nl) * Ss + half * 16;
  const __bf16* V3 = Vh + (size_t)(48 + nl) * Ss + half * 16;

  for (int kt = 0; kt < Ss; kt += 32) {
    // prefetch next iteration's K rows / V segments (global_prefetch path)
    if (kt + 32 < Ss) {
      __builtin_prefetch(Kh + (size_t)(kt + 32 + nl) * DK, 0, 0);
      __builtin_prefetch(Kh + (size_t)(kt + 48 + nl) * DK, 0, 0);
      __builtin_prefetch(V0 + kt + 32, 0, 0);
      __builtin_prefetch(V1 + kt + 32, 0, 0);
      __builtin_prefetch(V2 + kt + 32, 0, 0);
      __builtin_prefetch(V3 + kt + 32, 0, 0);
    }

    // ---- scores: two 16x16 tiles (keys kt..kt+15, kt+16..kt+31)
    v8f c0 = {}, c1 = {};
    {
      const __bf16* K0 = Kh + (size_t)(kt + nl) * DK;
      const __bf16* K1 = Kh + (size_t)(kt + 16 + nl) * DK;
      const int kb = half * 16;
      Frag16 kb00, kb01, kb10, kb11;
#pragma unroll
      for (int j = 0; j < 8; j++) {
        kb00.u[j] = *(const uint32_t*)(K0 + kb + 2 * j);
        kb01.u[j] = *(const uint32_t*)(K0 + 32 + kb + 2 * j);
        kb10.u[j] = *(const uint32_t*)(K1 + kb + 2 * j);
        kb11.u[j] = *(const uint32_t*)(K1 + 32 + kb + 2 * j);
      }
      c0 = WMMA_BF16(qa0.v, kb00.v, c0);
      c0 = WMMA_BF16(qa1.v, kb01.v, c0);
      c1 = WMMA_BF16(qa0.v, kb10.v, c1);
      c1 = WMMA_BF16(qa1.v, kb11.v, c1);
    }

    // ---- issue V fragment loads now; they overlap the softmax math below
    Frag16 vb0, vb1, vb2, vb3;
#pragma unroll
    for (int j = 0; j < 8; j++) {
      vb0.u[j] = *(const uint32_t*)(V0 + kt + 2 * j);
      vb1.u[j] = *(const uint32_t*)(V1 + kt + 2 * j);
      vb2.u[j] = *(const uint32_t*)(V2 + kt + 2 * j);
      vb3.u[j] = *(const uint32_t*)(V3 + kt + 2 * j);
    }

    // ---- mask words (1 u32 per row covers all 32 keys of this step)
    uint32_t mw[8];
#pragma unroll
    for (int r = 0; r < 8; r++) mw[r] = mrow[r][kt >> 5];

    // ---- scale, mask, online softmax (row stats via 16-lane xor reductions)
    float p0[8], p1[8], alpha[8];
#pragma unroll
    for (int r = 0; r < 8; r++) {
      float s0 = c0[r] * 0.125f;
      float s1 = c1[r] * 0.125f;
      if (!((mw[r] >> nl) & 1u))        s0 = -1e9f;
      if (!((mw[r] >> (16 + nl)) & 1u)) s1 = -1e9f;
      float vmax = fmaxf(s0, s1);
#pragma unroll
      for (int off = 1; off < 16; off <<= 1) vmax = fmaxf(vmax, __shfl_xor(vmax, off, 32));
      float mnew = fmaxf(mrun[r], vmax);
      float a  = __expf(mrun[r] - mnew);
      float e0 = __expf(s0 - mnew);
      float e1 = __expf(s1 - mnew);
      float rs = e0 + e1;
#pragma unroll
      for (int off = 1; off < 16; off <<= 1) rs += __shfl_xor(rs, off, 32);
      lrun[r] = lrun[r] * a + rs;
      mrun[r] = mnew;
      alpha[r] = a; p0[r] = e0; p1[r] = e1;
    }
#pragma unroll
    for (int r = 0; r < 8; r++) {
      o0[r] *= alpha[r]; o1[r] *= alpha[r]; o2[r] *= alpha[r]; o3[r] *= alpha[r];
    }

    // ---- P (16x32, f32 C-layout) -> bf16 -> LDS (col-major) -> A-fragment
#pragma unroll
    for (int r2 = 0; r2 < 4; r2++) {
      Pack2 a0, a1;
      a0.h[0] = f2bf(p0[2 * r2]); a0.h[1] = f2bf(p0[2 * r2 + 1]);
      a1.h[0] = f2bf(p1[2 * r2]); a1.h[1] = f2bf(p1[2 * r2 + 1]);
      int row0 = half * 8 + 2 * r2;
      *(uint32_t*)(Pw + nl * 16 + row0)        = a0.u;  // column nl
      *(uint32_t*)(Pw + (16 + nl) * 16 + row0) = a1.u;  // column 16+nl
    }
    Frag16 pa;
#pragma unroll
    for (int i = 0; i < 8; i++) {
      int k0 = half * 16 + 2 * i;
      pa.h[2 * i]     = Pw[k0 * 16 + nl];
      pa.h[2 * i + 1] = Pw[(k0 + 1) * 16 + nl];
    }

    // ---- PV: O(16x64) += P(16x32) x V(32x64)
    o0 = WMMA_BF16(pa.v, vb0.v, o0);
    o1 = WMMA_BF16(pa.v, vb1.v, o1);
    o2 = WMMA_BF16(pa.v, vb2.v, o2);
    o3 = WMMA_BF16(pa.v, vb3.v, o3);
  }

  // ---- normalize, write concat-head bf16 [B,S,D]
#pragma unroll
  for (int r = 0; r < 8; r++) {
    float inv = 1.0f / lrun[r];
    int q = qbase + half * 8 + r;
    __bf16* orow = attnOut + ((size_t)bi * Ss + q) * Dd + h * DK;
    orow[0  + nl] = f2bf(o0[r] * inv);
    orow[16 + nl] = f2bf(o1[r] * inv);
    orow[32 + nl] = f2bf(o2[r] * inv);
    orow[48 + nl] = f2bf(o3[r] * inv);
  }
}

// ---------------------------------------------------------------- launch
extern "C" void kernel_launch(void* const* d_in, const int* in_sizes, int n_in,
                              void* d_out, int out_size, void* d_ws, size_t ws_size,
                              hipStream_t stream) {
  const float* q    = (const float*)d_in[0];
  const float* k    = (const float*)d_in[1];
  const float* v    = (const float*)d_in[2];
  const int*   mask = (const int*)  d_in[3];
  const float* Wq   = (const float*)d_in[4];
  const float* bq   = (const float*)d_in[5];
  const float* Wk   = (const float*)d_in[6];
  const float* bk   = (const float*)d_in[7];
  const float* Wv   = (const float*)d_in[8];
  const float* bv   = (const float*)d_in[9];
  const float* Wo   = (const float*)d_in[10];
  const float* bo   = (const float*)d_in[11];
  float* out = (float*)d_out;

  char* ws = (char*)d_ws;
  size_t off = 0;
  auto alloc = [&](size_t bytes) {
    char* p = ws + off;
    off += (bytes + 255) & ~(size_t)255;
    return p;
  };
  const size_t XB = (size_t)Mrows * Dd * sizeof(__bf16);  // 8 MB
  const size_t WB = (size_t)Dd * Dd * sizeof(__bf16);     // 2 MB
  __bf16*   Xq   = (__bf16*)alloc(XB);
  __bf16*   Xk   = (__bf16*)alloc(XB);
  __bf16*   Xv   = (__bf16*)alloc(XB);
  __bf16*   WqT  = (__bf16*)alloc(WB);
  __bf16*   WkT  = (__bf16*)alloc(WB);
  __bf16*   WvT  = (__bf16*)alloc(WB);
  __bf16*   WoT  = (__bf16*)alloc(WB);
  __bf16*   Qbf  = (__bf16*)alloc(XB);   // [B,H,S,DK]
  __bf16*   Kbf  = (__bf16*)alloc(XB);   // [B,H,S,DK]
  __bf16*   Vtp  = (__bf16*)alloc(XB);   // [B,H,DK,S]
  __bf16*   Attn = (__bf16*)alloc(XB);   // [B,S,D]
  uint32_t* Mp   = (uint32_t*)alloc((size_t)Bb * Ss * MW * sizeof(uint32_t));  // 1 MB
  (void)ws_size; (void)in_sizes; (void)n_in; (void)out_size;

  const int n4 = Mrows * Dd / 4;  // 1M float4s
  mha_cvt_bf16<<<(n4 + 255) / 256, 256, 0, stream>>>(q, Xq, n4);
  mha_cvt_bf16<<<(n4 + 255) / 256, 256, 0, stream>>>(k, Xk, n4);
  mha_cvt_bf16<<<(n4 + 255) / 256, 256, 0, stream>>>(v, Xv, n4);

  dim3 tgrid(Dd / 32, Dd / 32), tblk(32, 8);
  mha_wtrans_bf16<<<tgrid, tblk, 0, stream>>>(Wq, WqT);
  mha_wtrans_bf16<<<tgrid, tblk, 0, stream>>>(Wk, WkT);
  mha_wtrans_bf16<<<tgrid, tblk, 0, stream>>>(Wv, WvT);
  mha_wtrans_bf16<<<tgrid, tblk, 0, stream>>>(Wo, WoT);

  const int nwords = Bb * Ss * MW;  // 262144
  mha_maskpack<<<(nwords + 255) / 256, 256, 0, stream>>>(mask, Mp, nwords);

  dim3 ggrid(Mrows / 64, Dd / 64);  // (64, 16)
  mha_gemm_bf16<<<ggrid, 128, 0, stream>>>(Xq, WqT, bq, Qbf, nullptr, 0);
  mha_gemm_bf16<<<ggrid, 128, 0, stream>>>(Xk, WkT, bk, Kbf, nullptr, 0);
  mha_gemm_bf16<<<ggrid, 128, 0, stream>>>(Xv, WvT, bv, Vtp, nullptr, 1);

  const int nTiles = Bb * Hh * (Ss / 16);  // 4096 waves
  mha_attn<<<nTiles / 4, 128, 0, stream>>>(Qbf, Kbf, Vtp, Mp, Attn);

  mha_gemm_bf16<<<ggrid, 128, 0, stream>>>(Attn, WoT, bo, nullptr, out, 2);
}
